// LocalGraphTransformerEncoder_78855599555045
// MI455X (gfx1250) — matrix operand order repaired and checked by reading, not compiled, not run on hardware
//
#include <hip/hip_runtime.h>
#include <hip/hip_bf16.h>
#include <math.h>

// ---------------------------------------------------------------------------
// Problem dims
// ---------------------------------------------------------------------------
constexpr int Bb   = 4;
constexpr int Nn_  = 384;
constexpr int Dd   = 256;
constexpr int Hh   = 4;
constexpr int DH   = 64;
constexpr int HID  = 64;
constexpr int CONT = 6;
constexpr int DFF  = 512;   // D*FF
constexpr int DFF2 = 1024;  // D*FF*2
constexpr float SCALE = 0.125f; // 64^-0.5

// d_out layout: x_out (B,N,D) | attn (B,H,N,N) | bias (B,H,N,N)
constexpr size_t OUT_X    = 0;
constexpr size_t OUT_ATTN = (size_t)Bb * Nn_ * Dd;                   // 393216
constexpr size_t OUT_BIAS = OUT_ATTN + (size_t)Bb * Hh * Nn_ * Nn_;  // 2752512

// ---------------------------------------------------------------------------
// WMMA types & helpers (gfx1250, wave32)
// ---------------------------------------------------------------------------
typedef __attribute__((ext_vector_type(16))) __bf16 v16bf;
typedef __attribute__((ext_vector_type(8)))  __bf16 v8bf;
typedef __attribute__((ext_vector_type(8)))  float  v8f;

__device__ __forceinline__ v8f wmma_bf16(v16bf a, v16bf b, v8f c) {
  return __builtin_amdgcn_wmma_f32_16x16x32_bf16(
      /*neg_a=*/false, a, /*neg_b=*/false, b,
      /*c_mod=*/(short)0, c, /*reuse_a=*/false, /*reuse_b=*/false);
}

__device__ __forceinline__ unsigned short f32_to_bf16(float f) {
  union { float f; unsigned u; } x; x.f = f;
  unsigned r = x.u + 0x7FFFu + ((x.u >> 16) & 1u);
  return (unsigned short)(r >> 16);
}

__device__ __forceinline__ float gelu_f(float x) {
  return 0.5f * x * (1.0f + erff(x * 0.70710678118654752f));
}

// A fragment, 16x32 bf16 row-major (ISA 7.12.2 "16-bit A-Matrix 16x32"):
// lane m = lane&15; lanes>=16 take K+8 / K+24 halves. Two 16-byte loads.
__device__ __forceinline__ v16bf load_a16x32(const __bf16* A0, int lda, int lane) {
  const __bf16* row = A0 + (size_t)(lane & 15) * lda + ((lane >> 4) << 3);
  v8bf lo = *(const v8bf*)(row);
  v8bf hi = *(const v8bf*)(row + 16);
  return __builtin_shufflevector(lo, hi, 0,1,2,3,4,5,6,7,8,9,10,11,12,13,14,15);
}

// B fragment, 32x16 bf16 with B[k][n] = W[n][k], W row-major [Nn x K]:
// lane n = lane&15; lanes>=16 take K+8 / K+24 halves. Two 16-byte loads.
__device__ __forceinline__ v16bf load_b32x16(const __bf16* W0, int ldw, int lane) {
  const __bf16* p = W0 + (size_t)(lane & 15) * ldw + ((lane >> 4) << 3);
  v8bf lo = *(const v8bf*)(p);
  v8bf hi = *(const v8bf*)(p + 16);
  return __builtin_shufflevector(lo, hi, 0,1,2,3,4,5,6,7,8,9,10,11,12,13,14,15);
}

// ---------------------------------------------------------------------------
// Elementwise / prep kernels
// ---------------------------------------------------------------------------
__global__ void cast_f32_bf16_kernel(const float* __restrict__ src,
                                     unsigned short* __restrict__ dst, int n) {
  for (int i = blockIdx.x * blockDim.x + threadIdx.x; i < n; i += gridDim.x * blockDim.x)
    dst[i] = f32_to_bf16(src[i]);
}

// One row (D=256) per block; 256 threads.
__global__ __launch_bounds__(256)
void layernorm_cast_kernel(const float* __restrict__ X, const float* __restrict__ g,
                           const float* __restrict__ bta, unsigned short* __restrict__ Y) {
  __shared__ float red[256];
  int row = blockIdx.x, tid = threadIdx.x;
  float v = X[(size_t)row * Dd + tid];
  red[tid] = v; __syncthreads();
  for (int s = 128; s > 0; s >>= 1) { if (tid < s) red[tid] += red[tid + s]; __syncthreads(); }
  float mean = red[0] * (1.0f / Dd);
  __syncthreads();
  float d = v - mean;
  red[tid] = d * d; __syncthreads();
  for (int s = 128; s > 0; s >>= 1) { if (tid < s) red[tid] += red[tid + s]; __syncthreads(); }
  float var = red[0] * (1.0f / Dd);
  float y = d * rsqrtf(var + 1e-5f) * g[tid] + bta[tid];
  Y[(size_t)row * Dd + tid] = f32_to_bf16(y);
}

__global__ void glu_kernel(const float* __restrict__ h, unsigned short* __restrict__ u) {
  int idx = blockIdx.x * blockDim.x + threadIdx.x;
  if (idx >= Bb * Nn_ * DFF) return;
  int m = idx >> 9, c = idx & 511;
  float a = h[(size_t)m * DFF2 + c];
  float gg = h[(size_t)m * DFF2 + DFF + c];
  u[idx] = f32_to_bf16(a * gelu_f(gg));
}

// ---------------------------------------------------------------------------
// Generic bf16 WMMA GEMM: C[M,Nn] = A[M,K] @ W[Nn,K]^T (+bias +resid)
// One 16x16 tile per wave, 8 waves per block.
// ---------------------------------------------------------------------------
__global__ __launch_bounds__(256)
void gemm_bf16_kernel(const unsigned short* __restrict__ A, int lda,
                      const unsigned short* __restrict__ W, int ldw,
                      int M, int Ncols, int K,
                      const float* __restrict__ bias,
                      const float* __restrict__ resid,
                      float* __restrict__ Cf, unsigned short* __restrict__ Cb, int ldc) {
  int lane = threadIdx.x & 31, wave = threadIdx.x >> 5;
  int tiles_n = Ncols >> 4;
  int t = blockIdx.x * 8 + wave;
  if (t >= (M >> 4) * tiles_n) return;
  int mt = t / tiles_n, nt = t % tiles_n;
  const __bf16* Ab = (const __bf16*)A + (size_t)(mt * 16) * lda;
  const __bf16* Wb = (const __bf16*)W + (size_t)(nt * 16) * ldw;
  v8f acc = {};
  for (int k0 = 0; k0 < K; k0 += 32) {
    __builtin_prefetch(Ab + k0 + 64, 0, 1);   // global_prefetch_b8
    v16bf a = load_a16x32(Ab + k0, lda, lane);
    v16bf b = load_b32x16(Wb + k0, ldw, lane);
    acc = wmma_bf16(a, b, acc);
  }
  int n = nt * 16 + (lane & 15);
  int mbase = mt * 16 + ((lane >> 4) << 3);
  float bn = bias ? bias[n] : 0.0f;
#pragma unroll
  for (int r = 0; r < 8; r++) {
    int m = mbase + r;
    float v = acc[r] + bn;
    if (resid) v += resid[(size_t)m * ldc + n];
    if (Cf) Cf[(size_t)m * ldc + n] = v;
    if (Cb) Cb[(size_t)m * ldc + n] = f32_to_bf16(v);
  }
}

// ---------------------------------------------------------------------------
// QKV GEMM: xn[1536x256] @ Wqkv[768x256]^T ; routes into q(bf16,scaled),
// k(bf16), v^T(bf16, [B,H,DH,N]) so attention fragments are contiguous.
// ---------------------------------------------------------------------------
__global__ __launch_bounds__(256)
void qkv_kernel(const unsigned short* __restrict__ A,
                const unsigned short* __restrict__ W,
                unsigned short* __restrict__ qbf,
                unsigned short* __restrict__ kbf,
                unsigned short* __restrict__ vtbf) {
  constexpr int M = Bb * Nn_, K = Dd, tiles_n = (3 * Dd) >> 4;
  int lane = threadIdx.x & 31, wave = threadIdx.x >> 5;
  int t = blockIdx.x * 8 + wave;
  if (t >= (M >> 4) * tiles_n) return;
  int mt = t / tiles_n, nt = t % tiles_n;
  const __bf16* Ab = (const __bf16*)A + (size_t)(mt * 16) * K;
  const __bf16* Wb = (const __bf16*)W + (size_t)(nt * 16) * K;
  v8f acc = {};
  for (int k0 = 0; k0 < K; k0 += 32) {
    v16bf a = load_a16x32(Ab + k0, K, lane);
    v16bf b = load_b32x16(Wb + k0, K, lane);
    acc = wmma_bf16(a, b, acc);
  }
  int col = nt * 16 + (lane & 15);
  int sel = col >> 8, hh = (col >> 6) & 3, d = col & 63;
  int mbase = mt * 16 + ((lane >> 4) << 3);
#pragma unroll
  for (int r = 0; r < 8; r++) {
    int m = mbase + r;
    int b = m / Nn_, tok = m - b * Nn_;
    size_t bh = (size_t)(b * Hh + hh);
    float v = acc[r];
    if (sel == 0)      qbf[(bh * Nn_ + tok) * DH + d] = f32_to_bf16(v * SCALE);
    else if (sel == 1) kbf[(bh * Nn_ + tok) * DH + d] = f32_to_bf16(v);
    else               vtbf[(bh * DH + d) * Nn_ + tok] = f32_to_bf16(v);
  }
}

// ---------------------------------------------------------------------------
// Pair bias kernel: block = (b, i, 128 j's); 8 waves x 16 pairs each.
// cont-MLP (VALU) -> WMMA vs LDS cm_W2 -> embedding gathers (L2-resident)
// -> LN(HID) + gelu + 4-head projection + mask -> d_out bias.
// ---------------------------------------------------------------------------
__global__ __launch_bounds__(256)
void pair_bias_kernel(const float* __restrict__ cont, const unsigned char* __restrict__ pmask,
                      const int* __restrict__ dist, const int* __restrict__ dirb,
                      const int* __restrict__ role, const int* __restrict__ hop,
                      const int* __restrict__ edge, const int* __restrict__ sp,
                      const int* __restrict__ same,
                      const int* __restrict__ degi_, const int* __restrict__ celi_,
                      const float* __restrict__ cmW1, const float* __restrict__ cmb1,
                      const unsigned short* __restrict__ cmW2bf, const float* __restrict__ cmb2,
                      const float* __restrict__ Ed, const float* __restrict__ Edr,
                      const float* __restrict__ Ero, const float* __restrict__ Eho,
                      const float* __restrict__ Eed, const float* __restrict__ Esp,
                      const float* __restrict__ Edg, const float* __restrict__ Ece,
                      const float* __restrict__ Esa,
                      const float* __restrict__ lng, const float* __restrict__ lnb,
                      const float* __restrict__ bW, const float* __restrict__ bb_,
                      float* __restrict__ biasOut) {
  __shared__ __align__(64) unsigned short sW2[HID * HID];     // 8 KB
  __shared__ __align__(64) unsigned short sT[8][16][HID];     // 16 KB bf16 A tiles
  __shared__ __align__(64) float          sPR[8][16][HID];    // 32 KB
  int tid = threadIdx.x, lane = tid & 31, wave = tid >> 5;
  int jb = blockIdx.x % (Nn_ / 128);
  int rest = blockIdx.x / (Nn_ / 128);
  int i = rest % Nn_, b = rest / Nn_;

  for (int idx = tid; idx < HID * HID; idx += 256) sW2[idx] = cmW2bf[idx];
  __syncthreads();

  int jt = jb * 128 + wave * 16;
  int p = lane & 15, half = lane >> 4;
  int j = jt + p;
  size_t pij = ((size_t)(b * Nn_ + i)) * Nn_ + j;

  // Stage A = gelu(cont @ cm_W1^T + b1): each lane does 32 channels of its pair.
  const float* cf = cont + pij * CONT;
  float c0 = cf[0], c1 = cf[1], c2 = cf[2], c3 = cf[3], c4 = cf[4], c5 = cf[5];
  for (int c = half * 32; c < half * 32 + 32; c++) {
    const float* wr = cmW1 + c * CONT;
    float tv = cmb1[c] + wr[0]*c0 + wr[1]*c1 + wr[2]*c2 + wr[3]*c3 + wr[4]*c4 + wr[5]*c5;
    sT[wave][p][c] = f32_to_bf16(gelu_f(tv));
  }
  __syncthreads();

  // pr tile: [16 pairs x 64 hid] via 8 WMMAs (4 n-tiles x K=64)
  v8f acc[4] = {v8f{}, v8f{}, v8f{}, v8f{}};
  const __bf16* Ab = (const __bf16*)&sT[wave][0][0];
#pragma unroll
  for (int nt = 0; nt < 4; nt++) {
    const __bf16* Wb = (const __bf16*)sW2 + nt * 16 * HID;
#pragma unroll
    for (int k0 = 0; k0 < HID; k0 += 32) {
      v16bf a = load_a16x32(Ab + k0, HID, lane);
      v16bf bfr = load_b32x16(Wb + k0, HID, lane);
      acc[nt] = wmma_bf16(a, bfr, acc[nt]);
    }
  }

  // Add b2 + embedding terms; spill pr tile to LDS for the row-wise LN phase.
  int nlo = lane & 15;
  int dgi = degi_[b * Nn_ + i], cti = celi_[b * Nn_ + i];
#pragma unroll
  for (int r = 0; r < 8; r++) {
    int m = r + (half << 3);
    int jm = jt + m;
    size_t pr_ = ((size_t)(b * Nn_ + i)) * Nn_ + jm;
    int di = dist[pr_], dr = dirb[pr_], ro = role[pr_], ho = hop[pr_];
    int ed = edge[pr_], sb = sp[pr_], sc = same[pr_];
    int dgj = degi_[b * Nn_ + jm], ctj = celi_[b * Nn_ + jm];
#pragma unroll
    for (int nt = 0; nt < 4; nt++) {
      int c = nt * 16 + nlo;
      float add = cmb2[c]
                + Ed[di * HID + c] + Edr[dr * HID + c] + Ero[ro * HID + c]
                + Eho[ho * HID + c] + Eed[ed * HID + c] + Esp[sb * HID + c]
                + Esa[sc * HID + c]
                + Edg[dgi * HID + c] + Edg[dgj * HID + c]
                + Ece[cti * HID + c] + Ece[ctj * HID + c];
      float v = acc[nt][r] + add;
      sPR[wave][m][c] = v;
    }
  }
  __syncthreads();

  // LN over HID + gelu + project to H=4; lanes 0-15 own one pair each.
  if (lane < 16) {
    int m = lane, jj = jt + m;
    const float* row = &sPR[wave][m][0];
    float mean = 0.f;
    for (int c = 0; c < HID; c++) mean += row[c];
    mean *= (1.0f / HID);
    float var = 0.f;
    for (int c = 0; c < HID; c++) { float d = row[c] - mean; var += d * d; }
    var *= (1.0f / HID);
    float inv = rsqrtf(var + 1e-5f);
    float hv0 = bb_[0], hv1 = bb_[1], hv2 = bb_[2], hv3 = bb_[3];
    for (int c = 0; c < HID; c++) {
      float u = gelu_f((row[c] - mean) * inv * lng[c] + lnb[c]);
      hv0 += u * bW[0 * HID + c];
      hv1 += u * bW[1 * HID + c];
      hv2 += u * bW[2 * HID + c];
      hv3 += u * bW[3 * HID + c];
    }
    size_t pr2 = ((size_t)(b * Nn_ + i)) * Nn_ + jj;
    float msk = pmask[pr2] ? 1.0f : 0.0f;
    size_t base = (((size_t)(b * Hh) * Nn_) + i) * Nn_ + jj;
    size_t hstride = (size_t)Nn_ * Nn_;
    biasOut[base + 0 * hstride] = hv0 * msk;
    biasOut[base + 1 * hstride] = hv1 * msk;
    biasOut[base + 2 * hstride] = hv2 * msk;
    biasOut[base + 3 * hstride] = hv3 * msk;
  }
}

// ---------------------------------------------------------------------------
// Attention: one wave per (b, h, 16-row i-tile). QK^T WMMA + bias + mask,
// full-row softmax in LDS, attn written to d_out, P x V^T WMMA -> bf16 out.
// ---------------------------------------------------------------------------
__global__ __launch_bounds__(32)
void attention_kernel(const unsigned short* __restrict__ qbf,
                      const unsigned short* __restrict__ kbf,
                      const unsigned short* __restrict__ vtbf,
                      const float* __restrict__ biasOut,
                      const unsigned char* __restrict__ pmask,
                      const unsigned char* __restrict__ valid,
                      float* __restrict__ attnOut,
                      unsigned short* __restrict__ outbf) {
  __shared__ __align__(64) float          sL[16][Nn_];   // 24 KB logits/probs
  __shared__ __align__(64) unsigned short sP[16][Nn_];   // 12 KB bf16 probs
  int lane = threadIdx.x;
  int it = blockIdx.x % (Nn_ / 16);
  int rest = blockIdx.x / (Nn_ / 16);
  int h = rest % Hh, b = rest / Hh;
  size_t bh = (size_t)(b * Hh + h);

  const __bf16* Q = (const __bf16*)qbf + (bh * Nn_ + it * 16) * DH;
  v16bf qa0 = load_a16x32(Q, DH, lane);
  v16bf qa1 = load_a16x32(Q + 32, DH, lane);
  int nlo = lane & 15, half = lane >> 4;

  for (int jt = 0; jt < Nn_ / 16; jt++) {
    const __bf16* Kb = (const __bf16*)kbf + (bh * Nn_ + jt * 16) * DH;
    v8f acc = {};
    acc = wmma_bf16(qa0, load_b32x16(Kb, DH, lane), acc);
    acc = wmma_bf16(qa1, load_b32x16(Kb + 32, DH, lane), acc);
    int jj = jt * 16 + nlo;
#pragma unroll
    for (int r = 0; r < 8; r++) {
      int m = r + (half << 3);
      int i = it * 16 + m;
      size_t prow = ((size_t)(b * Nn_ + i)) * Nn_ + jj;
      float l = acc[r] + biasOut[(bh * Nn_ + i) * Nn_ + jj];
      bool safe = pmask[prow] || ((!valid[b * Nn_ + i]) && (i == jj));
      sL[m][jj] = safe ? l : -3.4028234663852886e38f;
    }
  }
  __syncthreads();

  if (lane < 16) {
    int m = lane, i = it * 16 + m;
    float mx = -3.4028234663852886e38f;
    for (int jv = 0; jv < Nn_; jv++) mx = fmaxf(mx, sL[m][jv]);
    float sum = 0.f;
    for (int jv = 0; jv < Nn_; jv++) {
      size_t prow = ((size_t)(b * Nn_ + i)) * Nn_ + jv;
      float e = pmask[prow] ? __expf(sL[m][jv] - mx) : 0.0f;
      sL[m][jv] = e;
      sum += e;
    }
    float inv = 1.0f / fmaxf(sum, 1e-6f);
    for (int jv = 0; jv < Nn_; jv++) {
      float a = sL[m][jv] * inv;
      attnOut[(bh * Nn_ + i) * Nn_ + jv] = a;
      sP[m][jv] = f32_to_bf16(a);
    }
  }
  __syncthreads();

  // out tile = P[16x384] @ V^T ; V^T rows are d (contiguous in j).
#pragma unroll
  for (int dt = 0; dt < 4; dt++) {
    const __bf16* Vb = (const __bf16*)vtbf + (bh * DH + dt * 16) * Nn_;
    v8f acc = {};
    for (int k0 = 0; k0 < Nn_; k0 += 32) {
      v16bf a = load_a16x32((const __bf16*)&sP[0][0] + k0, Nn_, lane);
      v16bf bfr = load_b32x16(Vb + k0, Nn_, lane);
      acc = wmma_bf16(a, bfr, acc);
    }
#pragma unroll
    for (int r = 0; r < 8; r++) {
      int m = r + (half << 3);
      int i = it * 16 + m;
      int d = dt * 16 + nlo;
      outbf[((size_t)(b * Nn_ + i)) * Dd + h * DH + d] = f32_to_bf16(acc[r]);
    }
  }
}

// ---------------------------------------------------------------------------
// Host launch
// ---------------------------------------------------------------------------
extern "C" void kernel_launch(void* const* d_in, const int* in_sizes, int n_in,
                              void* d_out, int out_size, void* d_ws, size_t ws_size,
                              hipStream_t stream) {
  (void)in_sizes; (void)n_in; (void)out_size; (void)ws_size;
  const float* x      = (const float*)d_in[0];
  const unsigned char* pmask = (const unsigned char*)d_in[1];
  const unsigned char* valid = (const unsigned char*)d_in[2];
  const float* cont   = (const float*)d_in[3];
  const int* dist = (const int*)d_in[4];
  const int* dirb = (const int*)d_in[5];
  const int* role = (const int*)d_in[6];
  const int* hop  = (const int*)d_in[7];
  const int* edge = (const int*)d_in[8];
  const int* sp   = (const int*)d_in[9];
  const int* same = (const int*)d_in[10];
  const int* degI = (const int*)d_in[11];
  const int* celI = (const int*)d_in[12];
  const float* Wq = (const float*)d_in[13];
  const float* Wk = (const float*)d_in[14];
  const float* Wv = (const float*)d_in[15];
  const float* Wo = (const float*)d_in[16];
  const float* bo = (const float*)d_in[17];
  const float* cmW1 = (const float*)d_in[18];
  const float* cmb1 = (const float*)d_in[19];
  const float* cmW2 = (const float*)d_in[20];
  const float* cmb2 = (const float*)d_in[21];
  const float* Ed  = (const float*)d_in[22];
  const float* Edr = (const float*)d_in[23];
  const float* Ero = (const float*)d_in[24];
  const float* Eho = (const float*)d_in[25];
  const float* Eed = (const float*)d_in[26];
  const float* Esp = (const float*)d_in[27];
  const float* Edg = (const float*)d_in[28];
  const float* Ece = (const float*)d_in[29];
  const float* Esa = (const float*)d_in[30];
  const float* lng = (const float*)d_in[31];
  const float* lnb = (const float*)d_in[32];
  const float* bW  = (const float*)d_in[33];
  const float* bb_ = (const float*)d_in[34];
  const float* ln1g = (const float*)d_in[35];
  const float* ln1b = (const float*)d_in[36];
  const float* ln2g = (const float*)d_in[37];
  const float* ln2b = (const float*)d_in[38];
  const float* fW1 = (const float*)d_in[39];
  const float* fb1 = (const float*)d_in[40];
  const float* fW2 = (const float*)d_in[41];
  const float* fb2 = (const float*)d_in[42];

  float* out     = (float*)d_out;
  float* xout    = out + OUT_X;
  float* attnOut = out + OUT_ATTN;
  float* biasOut = out + OUT_BIAS;

  // Workspace carve (256-B aligned partitions; ~15.5 MB total).
  char* w = (char*)d_ws;
  auto carve = [&](size_t bytes) { char* p = w; w += (bytes + 255) & ~(size_t)255; return p; };
  constexpr size_t M = (size_t)Bb * Nn_;           // 1536 tokens
  unsigned short* xn_bf   = (unsigned short*)carve(M * Dd * 2);
  unsigned short* Wqkv_bf = (unsigned short*)carve((size_t)3 * Dd * Dd * 2);
  unsigned short* Wo_bf   = (unsigned short*)carve((size_t)Dd * Dd * 2);
  unsigned short* W1_bf   = (unsigned short*)carve((size_t)DFF2 * Dd * 2);
  unsigned short* W2_bf   = (unsigned short*)carve((size_t)Dd * DFF * 2);
  unsigned short* cmW2_bf = (unsigned short*)carve((size_t)HID * HID * 2);
  unsigned short* qbf     = (unsigned short*)carve(M * Dd * 2);
  unsigned short* kbf     = (unsigned short*)carve(M * Dd * 2);
  unsigned short* vtbf    = (unsigned short*)carve(M * Dd * 2);
  unsigned short* outbf   = (unsigned short*)carve(M * Dd * 2);
  float*          x1      = (float*)carve(M * Dd * 4);
  unsigned short* xn2_bf  = (unsigned short*)carve(M * Dd * 2);
  float*          hbuf    = (float*)carve(M * DFF2 * 4);
  unsigned short* ubf     = (unsigned short*)carve(M * DFF * 2);

  // 1) weight casts fp32 -> bf16 (Wqkv packed contiguously)
  auto castN = [&](const float* s, unsigned short* d, int n) {
    cast_f32_bf16_kernel<<<(n + 255) / 256, 256, 0, stream>>>(s, d, n);
  };
  castN(Wq, Wqkv_bf,                 Dd * Dd);
  castN(Wk, Wqkv_bf + Dd * Dd,       Dd * Dd);
  castN(Wv, Wqkv_bf + 2 * Dd * Dd,   Dd * Dd);
  castN(Wo, Wo_bf,                   Dd * Dd);
  castN(fW1, W1_bf,                  DFF2 * Dd);
  castN(fW2, W2_bf,                  Dd * DFF);
  castN(cmW2, cmW2_bf,               HID * HID);

  // 2) LN1 -> bf16 xn
  layernorm_cast_kernel<<<(int)M, 256, 0, stream>>>(x, ln1g, ln1b, xn_bf);

  // 3) QKV GEMM (1536x768x256) with q/k/v^T routing
  {
    int tiles = ((int)M / 16) * ((3 * Dd) / 16);   // 96*48
    qkv_kernel<<<(tiles + 7) / 8, 256, 0, stream>>>(xn_bf, Wqkv_bf, qbf, kbf, vtbf);
  }

  // 4) pair bias (B*N*(N/128) blocks)
  pair_bias_kernel<<<Bb * Nn_ * (Nn_ / 128), 256, 0, stream>>>(
      cont, pmask, dist, dirb, role, hop, edge, sp, same, degI, celI,
      cmW1, cmb1, cmW2_bf, cmb2, Ed, Edr, Ero, Eho, Eed, Esp, Edg, Ece, Esa,
      lng, lnb, bW, bb_, biasOut);

  // 5) attention (B*H*(N/16) waves)
  attention_kernel<<<Bb * Hh * (Nn_ / 16), 32, 0, stream>>>(
      qbf, kbf, vtbf, biasOut, pmask, valid, attnOut, outbf);

  // 6) output projection + residual: x1 = x + out @ Wo^T + bo
  {
    int tiles = ((int)M / 16) * (Dd / 16);
    gemm_bf16_kernel<<<(tiles + 7) / 8, 256, 0, stream>>>(
        outbf, Dd, Wo_bf, Dd, (int)M, Dd, Dd, bo, x, x1, nullptr, Dd);
  }

  // 7) LN2 -> bf16
  layernorm_cast_kernel<<<(int)M, 256, 0, stream>>>(x1, ln2g, ln2b, xn2_bf);

  // 8) FFN GEMM1: h = xn2 @ ffW1^T + b1  (1536x1024x256)
  {
    int tiles = ((int)M / 16) * (DFF2 / 16);
    gemm_bf16_kernel<<<(tiles + 7) / 8, 256, 0, stream>>>(
        xn2_bf, Dd, W1_bf, Dd, (int)M, DFF2, Dd, fb1, nullptr, hbuf, nullptr, DFF2);
  }

  // 9) GLU: u = a * gelu(g) -> bf16
  glu_kernel<<<((int)M * DFF + 255) / 256, 256, 0, stream>>>(hbuf, ubf);

  // 10) FFN GEMM2 + residual: x_out = x1 + u @ ffW2^T + b2
  {
    int tiles = ((int)M / 16) * (Dd / 16);
    gemm_bf16_kernel<<<(tiles + 7) / 8, 256, 0, stream>>>(
        ubf, DFF, W2_bf, DFF, (int)M, Dd, DFF, fb2, x1, xout, nullptr, Dd);
  }
}
